// Net_57690000720237
// MI455X (gfx1250) — compile-verified, matrix operand order checked by reading, not run on hardware
//
#include <hip/hip_runtime.h>
#include <hip/hip_bf16.h>

// ---------------------------------------------------------------------------
// MI455X (gfx1250, wave32) masked dual-expert SR net.
// Activations: NHWC f16 in workspace (halves traffic, makes WMMA fragments
// ds_load_b128-able). Accumulate f32 via V_WMMA_F32_16X16X32_F16.
// Interior input tiles loaded by the Tensor Data Mover (tensor_load_to_lds).
// ---------------------------------------------------------------------------

typedef __attribute__((ext_vector_type(16))) _Float16 v16h;
typedef __attribute__((ext_vector_type(8)))  _Float16 v8h;
typedef __attribute__((ext_vector_type(8)))  float    v8f;
typedef __attribute__((ext_vector_type(4)))  float    v4f;
typedef __attribute__((ext_vector_type(4)))  unsigned int v4u;
typedef __attribute__((ext_vector_type(8)))  int      v8i;
typedef __attribute__((ext_vector_type(4)))  int      v4i;

#define TW 16   // tile width  (pixels = WMMA M)
#define TH 8    // tile height (one wave32 per row -> 256 threads)

// ---------------------------------------------------------------------------
__global__ void __launch_bounds__(256) k_cvt_f16(const float* __restrict__ in,
                                                 _Float16* __restrict__ out, int n) {
  int i = blockIdx.x * 256 + threadIdx.x;
  if (i < n) out[i] = (_Float16)in[i];
}

// mask pass 1: 3x3 avg blur (count_include_pad=False), |x-blur| >= th
__global__ void __launch_bounds__(256) k_premask(const float* __restrict__ x,
                                                 const float* __restrict__ th,
                                                 unsigned char* __restrict__ pre,
                                                 int H, int W) {
  int p = blockIdx.x * 256 + threadIdx.x;
  if (p >= H * W) return;
  int img = blockIdx.y;
  int y = p / W, xx = p % W;
  const float* base = x + (size_t)img * H * W;
  float s = 0.f; int cnt = 0;
  #pragma unroll
  for (int dy = -1; dy <= 1; ++dy)
    #pragma unroll
    for (int dx = -1; dx <= 1; ++dx) {
      int yy = y + dy, xc = xx + dx;
      if (yy >= 0 && yy < H && xc >= 0 && xc < W) { s += base[yy * W + xc]; ++cnt; }
    }
  float blur = s / (float)cnt;
  pre[(size_t)img * H * W + p] = (fabsf(base[y * W + xx] - blur) >= th[0]) ? 1 : 0;
}

// mask pass 2: 3x3 dilate -> f16 {0,1}
__global__ void __launch_bounds__(256) k_dilate(const unsigned char* __restrict__ pre,
                                                _Float16* __restrict__ mask,
                                                int H, int W) {
  int p = blockIdx.x * 256 + threadIdx.x;
  if (p >= H * W) return;
  int img = blockIdx.y;
  int y = p / W, xx = p % W;
  const unsigned char* base = pre + (size_t)img * H * W;
  int m = 0;
  #pragma unroll
  for (int dy = -1; dy <= 1; ++dy)
    #pragma unroll
    for (int dx = -1; dx <= 1; ++dx) {
      int yy = y + dy, xc = xx + dx;
      if (yy >= 0 && yy < H && xc >= 0 && xc < W) m |= base[yy * W + xc];
    }
  mask[(size_t)img * H * W + p] = (_Float16)(m ? 1.f : 0.f);
}

// ---------------------------------------------------------------------------
// Pack conv weights (COUT,CIN,KS,KS) f32 -> B matrix [rows][KPAD] f16,
// K = tap-major, cin-minor, zero padded; rows >= nvalid are zero.
// ---------------------------------------------------------------------------
__global__ void __launch_bounds__(256) k_pack_w(const float* __restrict__ w,
                                                _Float16* __restrict__ out,
                                                int rows, int nvalid, int cin,
                                                int ks, int kpad) {
  int i = blockIdx.x * 256 + threadIdx.x;
  if (i >= rows * kpad) return;
  int n = i / kpad, kk = i % kpad;
  int ktot = ks * ks * cin;
  _Float16 v = (_Float16)0.f;
  if (kk < ktot && n < nvalid) {
    int tap = kk / cin, c = kk % cin;
    v = (_Float16)w[(n * cin + c) * ks * ks + tap];
  }
  out[i] = v;
}

// ---------------------------------------------------------------------------
// Fused dual-expert conv block over WMMA f16 (NHWC activations):
//   out = relu((conv(x,hw)+hb)*m + (conv1x1((conv(x,l1w)+l1b)*(1-m),l2w)+l2b)*(1-m))
// ---------------------------------------------------------------------------
template <int CIN, int COUT, int CM, int KS, int PAD>
__global__ void __launch_bounds__(256) k_conv_block(
    const _Float16* __restrict__ xin,   // [4][H][W][CIN]
    const _Float16* __restrict__ Whg,   // [COUT][KPAD] packed
    const _Float16* __restrict__ Wlg,   // [16][KPAD] packed (rows>=CM zero)
    const float* __restrict__ hb, const float* __restrict__ l1b,
    const float* __restrict__ l2w, const float* __restrict__ l2b,
    const _Float16* __restrict__ maskT, // [4][H][W]
    _Float16* __restrict__ xout,        // [4][H][W][COUT]
    int H, int W) {
  constexpr int KTOT   = KS * KS * CIN;
  constexpr int KPAD   = ((KTOT + 31) / 32) * 32;
  constexpr int KSTEPS = KPAD / 32;
  constexpr int NT     = (COUT + 15) / 16;
  constexpr int EH     = TH + KS - 1;
  constexpr int EW     = TW + KS - 1;

  __shared__ alignas(16) _Float16 xt[EH][EW][CIN];   // haloed NHWC tile
  __shared__ alignas(16) _Float16 msk[TH][TW];
  __shared__ alignas(16) float    outl[COUT][TH][TW];
  __shared__ alignas(16) float    x2l[CM][TH][TW];

  const int tid = threadIdx.x;
  const int x0  = blockIdx.x * TW;
  const int y0  = blockIdx.y * TH;
  const int img = blockIdx.z;

  __builtin_prefetch(Whg, 0, 1);  // global_prefetch_b8: B panel into GL2

  // ---- stage mask tile ----
  for (int i = tid; i < TH * TW; i += 256)
    msk[i / TW][i % TW] = maskT[(img * H + (y0 + i / TW)) * W + (x0 + i % TW)];

  // ---- stage haloed input tile: TDM for interior tiles, VMEM fallback ----
  const bool interior = (x0 >= PAD) && (y0 >= PAD) &&
                        (x0 + TW - 1 + PAD < W) && (y0 + TH - 1 + PAD < H);
  if (interior) {
    if (tid == 0) {
      // Tensor DMA descriptor (D#): 2-D tile EH x (EW*CIN) f16, row stride W*CIN
      unsigned long long ga = (unsigned long long)(uintptr_t)(
          xin + ((size_t)(img * H + (y0 - PAD)) * W + (x0 - PAD)) * CIN);
      unsigned ldsa = (unsigned)(uintptr_t)(void*)&xt[0][0][0];
      v4u g0 = { 1u,                                   // count=1, user mode
                 ldsa,                                 // lds_addr
                 (unsigned)(ga & 0xffffffffu),
                 (unsigned)((ga >> 32) & 0x01ffffffu) | 0x80000000u }; // type=2
      unsigned td0 = (unsigned)(W * CIN);              // tensor dim0 (elements)
      unsigned td1 = (unsigned)H;
      unsigned t0  = (unsigned)(EW * CIN);             // tile dim0
      unsigned t1  = (unsigned)EH;                     // tile dim1
      unsigned long long s0 = (unsigned long long)(W * CIN);
      v8i g1 = { (int)0x10000,                                 // data_size=1 (2B)
                 (int)((td0 & 0xffffu) << 16),                 // td0[15:0]
                 (int)(((td0 >> 16) & 0xffffu) | ((td1 & 0xffffu) << 16)),
                 (int)(((td1 >> 16) & 0xffffu) | ((t0 & 0xffffu) << 16)),
                 (int)(t1 & 0xffffu),                          // tile_dim1, dim2=0
                 (int)(s0 & 0xffffffffu),
                 (int)((s0 >> 32) & 0xffffu),                  // stride0 hi, stride1 lo=0
                 0 };
      v4i z4 = {0, 0, 0, 0};
      v8i z8 = {0, 0, 0, 0, 0, 0, 0, 0};
      __builtin_amdgcn_tensor_load_to_lds(g0, g1, z4, z4, z8, 0);
      __builtin_amdgcn_s_wait_tensorcnt(0);
    }
  } else {
    if constexpr (CIN % 8 == 0) {
      for (int i = tid; i < EH * EW * (CIN / 8); i += 256) {
        int r = i / (EW * (CIN / 8));
        int c = (i / (CIN / 8)) % EW;
        int g = i % (CIN / 8);
        int gy = y0 - PAD + r, gx = x0 - PAD + c;
        v8h v = {0, 0, 0, 0, 0, 0, 0, 0};
        if (gy >= 0 && gy < H && gx >= 0 && gx < W)
          v = *(const v8h*)&xin[((size_t)(img * H + gy) * W + gx) * CIN + g * 8];
        *(v8h*)&xt[r][c][g * 8] = v;
      }
    } else {
      for (int i = tid; i < EH * EW * CIN; i += 256) {
        int r = i / (EW * CIN);
        int c = (i / CIN) % EW;
        int ch = i % CIN;
        int gy = y0 - PAD + r, gx = x0 - PAD + c;
        _Float16 v = (_Float16)0.f;
        if (gy >= 0 && gy < H && gx >= 0 && gx < W)
          v = xin[((size_t)(img * H + gy) * W + gx) * CIN + ch];
        xt[r][c][ch] = v;
      }
    }
  }
  __syncthreads();

  const int wv   = tid >> 5;   // wave id = tile row
  const int lane = tid & 31;
  const int hi   = lane >> 4;  // half-wave -> K group / M group
  const int lp   = lane & 15;  // M (A) or N (B/C/D)

  // ---- A fragments (16-bit WMMA striping; two ds_load_b128 per fragment) ----
  v16h afrag[KSTEPS];
  #pragma unroll
  for (int ks = 0; ks < KSTEPS; ++ks) {
    if constexpr (CIN % 16 == 0) {
      v8h part[2];
      #pragma unroll
      for (int g = 0; g < 2; ++g) {
        int kk0 = ks * 32 + g * 16 + hi * 8;
        v8h v = {0, 0, 0, 0, 0, 0, 0, 0};
        if (kk0 < KTOT) {
          int tap = kk0 / CIN, c0 = kk0 % CIN;
          v = *(const v8h*)&xt[wv + tap / KS][lp + tap % KS][c0];
        }
        part[g] = v;
      }
      afrag[ks] = __builtin_shufflevector(part[0], part[1],
                                          0, 1, 2, 3, 4, 5, 6, 7,
                                          8, 9, 10, 11, 12, 13, 14, 15);
    } else {  // CIN==1 head layer: scalar gather (tiny K)
      v16h a;
      #pragma unroll
      for (int j = 0; j < 16; ++j) {
        int kl = (j < 8) ? (hi * 8 + j) : (16 + hi * 8 + (j - 8));
        int kk = ks * 32 + kl;
        _Float16 v = (_Float16)0.f;
        if (kk < KTOT) {
          int tap = kk / CIN;
          v = xt[wv + tap / KS][lp + tap % KS][0];
        }
        a[j] = v;
      }
      afrag[ks] = a;
    }
  }

  v8h mrow = *(const v8h*)&msk[wv][hi * 8];   // masks of this lane's 8 pixels

  // ---- hi expert: NT tiles of 16 couts (B straight from L2-resident pack) ----
  #pragma unroll
  for (int nt = 0; nt < NT; ++nt) {
    v8f acc = {0.f, 0.f, 0.f, 0.f, 0.f, 0.f, 0.f, 0.f};
    const int n = nt * 16 + lp;
    #pragma unroll
    for (int ks = 0; ks < KSTEPS; ++ks) {
      const _Float16* br = Whg + (size_t)n * KPAD + ks * 32 + hi * 8;
      v16h b = __builtin_shufflevector(*(const v8h*)br, *(const v8h*)(br + 16),
                                       0, 1, 2, 3, 4, 5, 6, 7,
                                       8, 9, 10, 11, 12, 13, 14, 15);
      acc = __builtin_amdgcn_wmma_f32_16x16x32_f16(false, afrag[ks], false, b,
                                                   (short)0, acc, false, false);
    }
    float bn = hb[n];
    v4f lo = { (acc[0] + bn) * (float)mrow[0], (acc[1] + bn) * (float)mrow[1],
               (acc[2] + bn) * (float)mrow[2], (acc[3] + bn) * (float)mrow[3] };
    v4f hi4 = { (acc[4] + bn) * (float)mrow[4], (acc[5] + bn) * (float)mrow[5],
                (acc[6] + bn) * (float)mrow[6], (acc[7] + bn) * (float)mrow[7] };
    *(v4f*)&outl[n][wv][hi * 8]     = lo;
    *(v4f*)&outl[n][wv][hi * 8 + 4] = hi4;
  }

  // ---- low-rank expert layer 1 (reuses A fragments) ----
  {
    v8f acc = {0.f, 0.f, 0.f, 0.f, 0.f, 0.f, 0.f, 0.f};
    #pragma unroll
    for (int ks = 0; ks < KSTEPS; ++ks) {
      const _Float16* br = Wlg + (size_t)lp * KPAD + ks * 32 + hi * 8;
      v16h b = __builtin_shufflevector(*(const v8h*)br, *(const v8h*)(br + 16),
                                       0, 1, 2, 3, 4, 5, 6, 7,
                                       8, 9, 10, 11, 12, 13, 14, 15);
      acc = __builtin_amdgcn_wmma_f32_16x16x32_f16(false, afrag[ks], false, b,
                                                   (short)0, acc, false, false);
    }
    if (lp < CM) {
      float bn = l1b[lp];
      v4f lo = { (acc[0] + bn) * (1.f - (float)mrow[0]),
                 (acc[1] + bn) * (1.f - (float)mrow[1]),
                 (acc[2] + bn) * (1.f - (float)mrow[2]),
                 (acc[3] + bn) * (1.f - (float)mrow[3]) };
      v4f hi4 = { (acc[4] + bn) * (1.f - (float)mrow[4]),
                  (acc[5] + bn) * (1.f - (float)mrow[5]),
                  (acc[6] + bn) * (1.f - (float)mrow[6]),
                  (acc[7] + bn) * (1.f - (float)mrow[7]) };
      *(v4f*)&x2l[lp][wv][hi * 8]     = lo;
      *(v4f*)&x2l[lp][wv][hi * 8 + 4] = hi4;
    }
  }
  __syncthreads();

  // ---- epilogue: rank-CM 1x1 expand (VALU), add, relu, NHWC v8h stores ----
  for (int p = tid; p < TH * TW; p += 256) {
    int yy = p / TW, xx = p % TW;
    int gy = y0 + yy, gx = x0 + xx;
    float mv = (float)msk[yy][xx];
    float inv = 1.f - mv;
    float x2v[CM];
    #pragma unroll
    for (int cm = 0; cm < CM; ++cm) x2v[cm] = x2l[cm][yy][xx];
    _Float16* op = xout + ((size_t)(img * H + gy) * W + gx) * COUT;
    #pragma unroll
    for (int g = 0; g < COUT / 8; ++g) {
      v8h pk;
      #pragma unroll
      for (int e = 0; e < 8; ++e) {
        int n = g * 8 + e;
        float s = l2b[n];
        #pragma unroll
        for (int cm = 0; cm < CM; ++cm) s += l2w[n * CM + cm] * x2v[cm];
        float v = outl[n][yy][xx] + s * inv;
        pk[e] = (_Float16)fmaxf(v, 0.f);
      }
      *(v8h*)(op + g * 8) = pk;
    }
  }
}

// ---------------------------------------------------------------------------
// 1x1 conv 32 -> 8 (NHWC), unmasked projection before the tconv tail
// ---------------------------------------------------------------------------
__global__ void __launch_bounds__(256) k_l1x(const _Float16* __restrict__ xin,
                                             const float* __restrict__ w,   // [8][32]
                                             const float* __restrict__ b,   // [8]
                                             _Float16* __restrict__ xout,
                                             int H, int W) {
  __shared__ float sw[8 * 32];
  __shared__ float sb[8];
  int tid = threadIdx.x;
  for (int i = tid; i < 8 * 32; i += 256) sw[i] = w[i];
  if (tid < 8) sb[tid] = b[tid];
  __syncthreads();
  int p = blockIdx.x * 256 + tid;
  if (p >= H * W) return;
  int img = blockIdx.y;
  const _Float16* px = xin + ((size_t)img * H * W + p) * 32;
  float acc[8];
  #pragma unroll
  for (int o = 0; o < 8; ++o) acc[o] = sb[o];
  #pragma unroll
  for (int g = 0; g < 4; ++g) {
    v8h xv = *(const v8h*)(px + g * 8);
    #pragma unroll
    for (int e = 0; e < 8; ++e) {
      float xe = (float)xv[e];
      #pragma unroll
      for (int o = 0; o < 8; ++o) acc[o] += xe * sw[o * 32 + g * 8 + e];
    }
  }
  v8h pk;
  #pragma unroll
  for (int o = 0; o < 8; ++o) pk[o] = (_Float16)acc[o];
  *(v8h*)(xout + ((size_t)img * H * W + p) * 8) = pk;
}

// ---------------------------------------------------------------------------
// Tail: out = (tconv9x9s2(x32,hw)+hb)*mu + (tconv9x9s2(x8,l2w)+l2b)*(1-mu)
// cout==1 -> VALU FMA (WMMA N=16 would be 94% idle); NHWC channel-vector loads.
// ---------------------------------------------------------------------------
__global__ void __launch_bounds__(256) k_last(const _Float16* __restrict__ x32,
                                              const _Float16* __restrict__ x8,
                                              const float* __restrict__ hw,  // [32][1][9][9]
                                              const float* __restrict__ hb,
                                              const float* __restrict__ l2w, // [8][1][9][9]
                                              const float* __restrict__ l2b,
                                              const _Float16* __restrict__ maskT,
                                              float* __restrict__ out,
                                              int H, int W, int OH, int OW) {
  __shared__ float s_hw[81 * 32];   // [tap][cin]
  __shared__ float s_l2[81 * 8];    // [tap][cin]
  int tid = threadIdx.x;
  for (int i = tid; i < 81 * 32; i += 256) { int wt = i >> 5, c = i & 31; s_hw[i] = hw[c * 81 + wt]; }
  for (int i = tid; i < 81 * 8; i += 256)  { int wt = i >> 3, c = i & 7;  s_l2[i] = l2w[c * 81 + wt]; }
  __syncthreads();

  int ox = blockIdx.x * 16 + (tid & 15);
  int oy = blockIdx.y * 16 + (tid >> 4);
  int img = blockIdx.z;
  float mv  = (float)maskT[(img * H + (oy >> 1)) * W + (ox >> 1)];
  float inv = 1.f - mv;

  float acc1 = 0.f, acc2 = 0.f;
  #pragma unroll
  for (int ky = 0; ky < 9; ++ky) {
    int ay = oy - 4 + ky;
    if (ay < 0 || (ay & 1) || (ay >> 1) >= H) continue;
    int iy = ay >> 1;
    #pragma unroll
    for (int kx = 0; kx < 9; ++kx) {
      int ax = ox - 4 + kx;
      if (ax < 0 || (ax & 1) || (ax >> 1) >= W) continue;
      int ix = ax >> 1;
      int wt = (8 - ky) * 9 + (8 - kx);          // rot180 (torch tconv weight)
      if (mv != 0.f) {
        const _Float16* px = x32 + ((size_t)(img * H + iy) * W + ix) * 32;
        #pragma unroll
        for (int g = 0; g < 4; ++g) {
          v8h xv = *(const v8h*)(px + g * 8);
          #pragma unroll
          for (int e = 0; e < 8; ++e) acc1 += (float)xv[e] * s_hw[wt * 32 + g * 8 + e];
        }
      }
      if (inv != 0.f) {
        const _Float16* q = x8 + ((size_t)(img * H + iy) * W + ix) * 8;
        v8h xv = *(const v8h*)q;
        #pragma unroll
        for (int e = 0; e < 8; ++e) acc2 += (float)xv[e] * s_l2[wt * 8 + e];
      }
    }
  }
  out[((size_t)img * OH + oy) * OW + ox] = (acc1 + hb[0]) * mv + (acc2 + l2b[0]) * inv;
}

// ---------------------------------------------------------------------------
// launch
// ---------------------------------------------------------------------------
extern "C" void kernel_launch(void* const* d_in, const int* in_sizes, int n_in,
                              void* d_out, int out_size, void* d_ws, size_t ws_size,
                              hipStream_t stream) {
  (void)in_sizes; (void)n_in; (void)out_size; (void)ws_size;
  const float* inx = (const float*)d_in[0];
  const float* th  = (const float*)d_in[1];
  auto P = [&](int i) { return (const float*)d_in[i]; };
  // d_in: 0 inx, 1 th, then {hw,hb,l1w,l1b,l2w,l2b}: 2..7 first, 8..13 reduction,
  // 14..19 mid1, 20..25 mid2, 26..31 mid3, 32..37 mid4, 38..43 expansion, 44..49 last.

  const int N = 4, H = 512, W = 512, OH = 1024, OW = 1024;
  char* ws = (char*)d_ws;
  size_t off = 0;
  auto alloc = [&](size_t bytes) {
    void* p = ws + off;
    off = (off + bytes + 255) & ~(size_t)255;
    return p;
  };
  _Float16*      mask = (_Float16*)alloc((size_t)N * H * W * 2);
  unsigned char* pre  = (unsigned char*)alloc((size_t)N * H * W);
  _Float16*      actA = (_Float16*)alloc((size_t)N * H * W * 1 * 2);   // NHWC
  _Float16*      actB = (_Float16*)alloc((size_t)N * H * W * 32 * 2);
  _Float16*      actC = (_Float16*)alloc((size_t)N * H * W * 16 * 2);
  _Float16*      actD = (_Float16*)alloc((size_t)N * H * W * 16 * 2);
  _Float16*      actE = (_Float16*)alloc((size_t)N * H * W * 8 * 2);
  // packed B panels [rows][KPAD] f16
  _Float16* Wfh = (_Float16*)alloc(32 * 32 * 2);
  _Float16* Wfl = (_Float16*)alloc(16 * 32 * 2);
  _Float16* Wrh = (_Float16*)alloc(16 * 32 * 2);
  _Float16* Wrl = (_Float16*)alloc(16 * 32 * 2);
  _Float16* Wmh[4]; _Float16* Wml[4];
  for (int m = 0; m < 4; ++m) {
    Wmh[m] = (_Float16*)alloc(16 * 160 * 2);
    Wml[m] = (_Float16*)alloc(16 * 160 * 2);
  }
  _Float16* Weh = (_Float16*)alloc(32 * 32 * 2);
  _Float16* Wel = (_Float16*)alloc(16 * 32 * 2);

  auto pack = [&](const float* w, _Float16* dst, int rows, int nvalid, int cin,
                  int ks, int kpad) {
    int n = rows * kpad;
    k_pack_w<<<(n + 255) / 256, 256, 0, stream>>>(w, dst, rows, nvalid, cin, ks, kpad);
  };
  pack(P(2),  Wfh, 32, 32, 1, 5, 32);    // first hw
  pack(P(4),  Wfl, 16, 8,  1, 5, 32);    // first l1w
  pack(P(8),  Wrh, 16, 16, 32, 1, 32);   // reduction hw
  pack(P(10), Wrl, 16, 4,  32, 1, 32);   // reduction l1w
  for (int m = 0; m < 4; ++m) {
    pack(P(14 + 6 * m), Wmh[m], 16, 16, 16, 3, 160);
    pack(P(16 + 6 * m), Wml[m], 16, 4,  16, 3, 160);
  }
  pack(P(38), Weh, 32, 32, 16, 1, 32);   // expansion hw
  pack(P(40), Wel, 16, 8,  16, 1, 32);   // expansion l1w

  int npix = N * H * W;
  k_cvt_f16<<<(npix + 255) / 256, 256, 0, stream>>>(inx, actA, npix);
  dim3 mg((H * W + 255) / 256, N);
  k_premask<<<mg, 256, 0, stream>>>(inx, th, pre, H, W);
  k_dilate<<<mg, 256, 0, stream>>>(pre, mask, H, W);

  dim3 cg(W / TW, H / TH, N);
  k_conv_block<1, 32, 8, 5, 2><<<cg, 256, 0, stream>>>(
      actA, Wfh, Wfl, P(3), P(5), P(6), P(7), mask, actB, H, W);
  k_conv_block<32, 16, 4, 1, 0><<<cg, 256, 0, stream>>>(
      actB, Wrh, Wrl, P(9), P(11), P(12), P(13), mask, actC, H, W);
  k_conv_block<16, 16, 4, 3, 1><<<cg, 256, 0, stream>>>(
      actC, Wmh[0], Wml[0], P(15), P(17), P(18), P(19), mask, actD, H, W);
  k_conv_block<16, 16, 4, 3, 1><<<cg, 256, 0, stream>>>(
      actD, Wmh[1], Wml[1], P(21), P(23), P(24), P(25), mask, actC, H, W);
  k_conv_block<16, 16, 4, 3, 1><<<cg, 256, 0, stream>>>(
      actC, Wmh[2], Wml[2], P(27), P(29), P(30), P(31), mask, actD, H, W);
  k_conv_block<16, 16, 4, 3, 1><<<cg, 256, 0, stream>>>(
      actD, Wmh[3], Wml[3], P(33), P(35), P(36), P(37), mask, actC, H, W);
  k_conv_block<16, 32, 8, 1, 0><<<cg, 256, 0, stream>>>(
      actC, Weh, Wel, P(39), P(41), P(42), P(43), mask, actB, H, W);

  k_l1x<<<mg, 256, 0, stream>>>(actB, P(46), P(47), actE, H, W);
  dim3 lg(OW / 16, OH / 16, N);
  k_last<<<lg, 256, 0, stream>>>(actB, actE, P(44), P(45), P(48), P(49), mask,
                                 (float*)d_out, H, W, OH, OW);
}
